// RecurrentBase_2250562863639
// MI455X (gfx1250) — compile-verified, matrix operand order checked by reading, not run on hardware
//
#include <hip/hip_runtime.h>
#include <hip/hip_bf16.h>
#include <stdint.h>

// ---------------------------------------------------------------------------
// GRU scan for MI455X (gfx1250, wave32, WMMA).
//
// ~206 GFLOP of GEMM vs ~150 MB HBM traffic -> compute bound -> bf16 WMMA
// (v_wmma_f32_16x16x32_bf16, f32 accumulate) for both the time-parallel input
// GEMM (GI = x @ W_ih^T + b_ih) and the 64 sequential recurrent GEMMs (fused
// with the gate math).
//
// v2: B-operand tiles are staged global->LDS (double buffered, shared by all
// 8 waves of the block: 8x less global B traffic), the K loop is software-
// pipelined (next fills + next A fragment issued before computing on the
// current buffer), and the CDNA5 async global->LDS copy path is used when the
// toolchain exposes it (s_wait_asynccnt before the barrier); otherwise
// fallback to load + ds_store which still pipelines across iterations.
//
// v2.1: fix async builtin argument types: it takes typed v4i pointers with
// explicit address_space(1)/(3) qualifiers (learned from the clang
// diagnostic), not void*.
// ---------------------------------------------------------------------------

#define T_STEPS 64
#define N_BATCH 256
#define IN_DIM  1024
#define H_DIM   1024
#define OC_DIM  (3 * H_DIM)
#define BROW    40   // padded LDS row stride (elements) to spread banks

typedef __attribute__((ext_vector_type(16))) __bf16 v16bf;
typedef __attribute__((ext_vector_type(8)))  float  v8f;

union Frag {
    v16bf v;
    unsigned int u[8];
};

// A/B fragment for v_wmma_f32_16x16x32_bf16 (ISA 7.12.2, 16-bit 16x32):
// lane L<16 holds row/col L, elements 0..7 = K kb..kb+7, elements 8..15 =
// K kb+16..kb+23; lanes 16..31 hold the +8 K offsets. Caller passes
// p = base + row*stride + kb + (lane>>4)*8; we load 16B at p and p+16 elems.
__device__ __forceinline__ void load_frag(Frag& f, const unsigned short* p) {
    *reinterpret_cast<uint4*>(&f.u[0]) = *reinterpret_cast<const uint4*>(p);
    *reinterpret_cast<uint4*>(&f.u[4]) = *reinterpret_cast<const uint4*>(p + 16);
}

__device__ __forceinline__ unsigned short f2bf(float f) {
    unsigned int u = __float_as_uint(f);
    u += 0x7FFFu + ((u >> 16) & 1u);   // round-to-nearest-even
    return (unsigned short)(u >> 16);
}

__device__ __forceinline__ float sigmoidf(float x) {
    return 1.0f / (1.0f + __expf(-x));
}

#define WMMA_BF16(A, B, C) \
    __builtin_amdgcn_wmma_f32_16x16x32_bf16(false, (A), false, (B), (short)0, (C), false, false)

// ---- async global->LDS (CDNA5) with fallback ------------------------------
#if __has_builtin(__builtin_amdgcn_global_load_async_to_lds_b128)
#define HAVE_ASYNC_LDS 1
typedef int v4i_ __attribute__((vector_size(16)));
typedef __attribute__((address_space(1))) v4i_* as1_v4i;
typedef __attribute__((address_space(3))) v4i_* as3_v4i;

__device__ __forceinline__ void async_copy16(const unsigned short* gsrc,
                                             unsigned short* ldst) {
    // AS1 value == flat value for global; AS3 value == flat[31:0] (ISA
    // aperture rule: LDS address is the low 32 bits).
    __builtin_amdgcn_global_load_async_to_lds_b128(
        (as1_v4i)(uintptr_t)gsrc,
        (as3_v4i)(unsigned int)(uintptr_t)ldst,
        0, 0);
}
__device__ __forceinline__ void wait_async0() {
#if __has_builtin(__builtin_amdgcn_s_wait_asynccnt)
    __builtin_amdgcn_s_wait_asynccnt(0);
#else
    asm volatile("s_wait_asynccnt 0x0" ::: "memory");
#endif
}
#define FILL_ISSUE(gsrc, ldst, frv)  async_copy16((gsrc), (ldst))
#define FILL_COMMIT(ldst, frv)       do { } while (0)
#define FILL_WAIT()                  wait_async0()
#else
#define FILL_ISSUE(gsrc, ldst, frv)  (frv) = *reinterpret_cast<const uint4*>(gsrc)
#define FILL_COMMIT(ldst, frv)       *reinterpret_cast<uint4*>(ldst) = (frv)
#define FILL_WAIT()                  do { } while (0)
#endif

// ---------------------------------------------------------------------------
// f32 -> bf16 conversion (x, W_ih, W_hh)
// ---------------------------------------------------------------------------
__global__ void __launch_bounds__(256) cvt_bf16_kernel(
    const float* __restrict__ in, unsigned short* __restrict__ out, int n) {
    int i = blockIdx.x * 256 + threadIdx.x;
    if (i < n) out[i] = f2bf(in[i]);
}

// ---------------------------------------------------------------------------
// hm0 = hxs * masks[0]  (f32 + bf16 copies)
// ---------------------------------------------------------------------------
__global__ void __launch_bounds__(256) init_hm_kernel(
    const float* __restrict__ hxs, const float* __restrict__ masks0,
    float* __restrict__ hmf, unsigned short* __restrict__ hmb) {
    int i = blockIdx.x * 256 + threadIdx.x;           // over N*H
    float v = hxs[i] * masks0[i >> 10];               // H_DIM == 1024
    hmf[i] = v;
    hmb[i] = f2bf(v);
}

// ---------------------------------------------------------------------------
// GI = x_bf16 @ W_ih_bf16^T + b_ih    [16384 x 3072], K = 1024
// Block = 128 rows x 64 cols (8 waves, one 16-row tile each, sharing 4 B
// tiles staged in LDS). 128 rowGroups * 48 colQuads = 6144 blocks.
// ---------------------------------------------------------------------------
__global__ void __launch_bounds__(256) gi_gemm_kernel(
    const unsigned short* __restrict__ xb,    // [16384,1024] bf16
    const unsigned short* __restrict__ wb,    // [3072,1024]  bf16
    const float* __restrict__ bias,           // [3072]
    float* __restrict__ gi)                   // [16384,3072] f32
{
    __shared__ unsigned short smB[2][64 * BROW];      // 2 x 5120B

    const int K = IN_DIM;
    const int tid  = threadIdx.x;
    const int wv   = tid >> 5;
    const int lane = tid & 31;
    const int l15  = lane & 15;
    const int hs   = lane >> 4;

    const int rowGroup = blockIdx.x / 48;
    const int colQuad  = blockIdx.x - rowGroup * 48;
    const int colBase  = colQuad * 64;
    const int rowTile  = rowGroup * 8 + wv;

    // fill mapping: 256 threads x 16B = 4KB = 4 tiles * 16 rows * 32 K bf16
    const int f_tile = tid >> 6;
    const int f_wrow = (tid >> 2) & 15;
    const int f_kq   = tid & 3;
    const unsigned short* fsrc =
        wb + (size_t)(colBase + f_tile * 16 + f_wrow) * K + f_kq * 8;
    unsigned short* fdst[2];
    fdst[0] = &smB[0][(f_tile * 16 + f_wrow) * BROW + f_kq * 8];
    fdst[1] = &smB[1][(f_tile * 16 + f_wrow) * BROW + f_kq * 8];

    const unsigned short* arow = xb + (size_t)(rowTile * 16 + l15) * K + hs * 8;

    v8f acc0 = {}, acc1 = {}, acc2 = {}, acc3 = {};
    Frag a[2];

    // ---- preload: buffer 0 + A(0)
    {
        uint4 fr; (void)fr;
        FILL_ISSUE(fsrc, fdst[0], fr);
        load_frag(a[0], arow);
        FILL_COMMIT(fdst[0], fr);
        FILL_WAIT();
        __syncthreads();
    }

#pragma unroll 2
    for (int i = 0; i < 32; ++i) {
        const int cur = i & 1;
        const int kb  = i << 5;
        const bool more = (i < 31);
        uint4 fr; (void)fr;
        if (more) {
            FILL_ISSUE(fsrc + kb + 32, fdst[cur ^ 1], fr);   // next B chunk
            load_frag(a[cur ^ 1], arow + kb + 32);           // next A frag
        }
        // compute on current LDS buffer
        Frag b0, b1, b2, b3;
        const unsigned short* bp = &smB[cur][l15 * BROW + hs * 8];
        load_frag(b0, bp);
        load_frag(b1, bp + 16 * BROW);
        load_frag(b2, bp + 32 * BROW);
        load_frag(b3, bp + 48 * BROW);
        acc0 = WMMA_BF16(a[cur].v, b0.v, acc0);
        acc1 = WMMA_BF16(a[cur].v, b1.v, acc1);
        acc2 = WMMA_BF16(a[cur].v, b2.v, acc2);
        acc3 = WMMA_BF16(a[cur].v, b3.v, acc3);
        if (more) {
            FILL_COMMIT(fdst[cur ^ 1], fr);
            FILL_WAIT();
            __syncthreads();
        }
    }

    // D layout: lane<16 -> M = e; lanes 16..31 -> M = 8+e.
    const int rbase = rowTile * 16 + hs * 8;
    const int c0 = colBase + l15;
    float b0 = bias[c0], b1 = bias[c0 + 16], b2 = bias[c0 + 32], b3 = bias[c0 + 48];
#pragma unroll
    for (int e = 0; e < 8; ++e) {
        size_t o = (size_t)(rbase + e) * OC_DIM + c0;
        gi[o]      = acc0[e] + b0;
        gi[o + 16] = acc1[e] + b1;
        gi[o + 32] = acc2[e] + b2;
        gi[o + 48] = acc3[e] + b3;
    }
}

// ---------------------------------------------------------------------------
// One GRU time step, fully fused:
//   GH(r,z,n) = hm_bf16 @ W_hh_bf16^T ; gates ; outs[t] ; hm_next ping-pong.
// Block = 8 row-tiles x 1 col-strip: 8 waves share the 3 r/z/n B tiles
// staged in LDS. 2 rowGroups * 64 colStrips = 128 blocks.
// ---------------------------------------------------------------------------
__global__ void __launch_bounds__(256) gru_step_kernel(
    const float* __restrict__ gi_t,            // [256,3072] (GI rows for step t)
    const unsigned short* __restrict__ hmb,    // [256,1024] bf16 masked h (in)
    const float* __restrict__ hmf,             // [256,1024] f32  masked h (in)
    const unsigned short* __restrict__ whh,    // [3072,1024] bf16
    const float* __restrict__ bhh,             // [3072]
    const float* __restrict__ mnext,           // [256] masks for step t+1
    float* __restrict__ out_h,                 // [256,1024] -> outs[t]
    float* __restrict__ hlast,                 // [256,1024] -> h_last (if last)
    float* __restrict__ hmf_out,               // ping-pong out (f32)
    unsigned short* __restrict__ hmb_out,      // ping-pong out (bf16)
    int is_last)
{
    __shared__ unsigned short smB[2][48 * BROW];      // 2 x 3840B

    const int K = H_DIM;
    const int tid  = threadIdx.x;
    const int wv   = tid >> 5;
    const int lane = tid & 31;
    const int l15  = lane & 15;
    const int hs   = lane >> 4;

    const int rowGroup = blockIdx.x >> 6;             // 0..1
    const int colStrip = blockIdx.x & 63;             // 0..63
    const int rowTile  = rowGroup * 8 + wv;           // 0..15

    // fill mapping: 192 threads x 16B = 3KB = 3 gates * 16 rows * 32 K bf16
    const bool filler = (tid < 192);
    const int f_gate = tid >> 6;
    const int f_wrow = (tid >> 2) & 15;
    const int f_kq   = tid & 3;
    const unsigned short* fsrc =
        whh + (size_t)(f_gate * H_DIM + colStrip * 16 + f_wrow) * K + f_kq * 8;
    unsigned short* fdst[2];
    fdst[0] = &smB[0][(f_gate * 16 + f_wrow) * BROW + f_kq * 8];
    fdst[1] = &smB[1][(f_gate * 16 + f_wrow) * BROW + f_kq * 8];

    const unsigned short* arow = hmb + (size_t)(rowTile * 16 + l15) * K + hs * 8;

    v8f ar = {}, az = {}, an = {};
    Frag a[2];

    // ---- preload
    {
        uint4 fr; (void)fr;
        if (filler) { FILL_ISSUE(fsrc, fdst[0], fr); }
        load_frag(a[0], arow);
        if (filler) { FILL_COMMIT(fdst[0], fr); }
        FILL_WAIT();
        __syncthreads();
    }

#pragma unroll 2
    for (int i = 0; i < 32; ++i) {
        const int cur = i & 1;
        const int kb  = i << 5;
        const bool more = (i < 31);
        uint4 fr; (void)fr;
        if (more) {
            if (filler) { FILL_ISSUE(fsrc + kb + 32, fdst[cur ^ 1], fr); }
            load_frag(a[cur ^ 1], arow + kb + 32);
        }
        Frag br, bz, bn;
        const unsigned short* bp = &smB[cur][l15 * BROW + hs * 8];
        load_frag(br, bp);
        load_frag(bz, bp + 16 * BROW);
        load_frag(bn, bp + 32 * BROW);
        ar = WMMA_BF16(a[cur].v, br.v, ar);
        az = WMMA_BF16(a[cur].v, bz.v, az);
        an = WMMA_BF16(a[cur].v, bn.v, an);
        if (more) {
            if (filler) { FILL_COMMIT(fdst[cur ^ 1], fr); }
            FILL_WAIT();
            __syncthreads();
        }
    }

    const int col = colStrip * 16 + l15;
    const float bgr = bhh[col], bgz = bhh[col + H_DIM], bgn = bhh[col + 2 * H_DIM];
    const int rbase = rowTile * 16 + hs * 8;

#pragma unroll
    for (int e = 0; e < 8; ++e) {
        int row = rbase + e;
        size_t g = (size_t)row * OC_DIM + col;
        float r = sigmoidf(gi_t[g] + ar[e] + bgr);
        float z = sigmoidf(gi_t[g + H_DIM] + az[e] + bgz);
        float n = tanhf(gi_t[g + 2 * H_DIM] + r * (an[e] + bgn));
        size_t ho = (size_t)row * H_DIM + col;
        float hm = hmf[ho];
        float hnew = (1.0f - z) * n + z * hm;
        out_h[ho] = hnew;
        if (is_last) {
            hlast[ho] = hnew;
        } else {
            float hmn = hnew * mnext[row];
            hmf_out[ho] = hmn;
            hmb_out[ho] = f2bf(hmn);
        }
    }
}

// ---------------------------------------------------------------------------
// Launch. Inputs (setup_inputs order):
//   0 x[T*N,IN] 1 hxs[N,H] 2 masks[T*N,1] 3 W_ih[3H,IN] 4 W_hh[3H,H]
//   5 b_ih[3H]  6 b_hh[3H]
// Output: outs[T*N,H] ++ h_last[N,H]  (f32)
// Workspace (~251 MB): x_bf16 | Wih_bf16 | Whh_bf16 | GI | hm_f32 x2 | hm_bf16 x2
// ---------------------------------------------------------------------------
extern "C" void kernel_launch(void* const* d_in, const int* in_sizes, int n_in,
                              void* d_out, int out_size, void* d_ws, size_t ws_size,
                              hipStream_t stream) {
    (void)in_sizes; (void)n_in; (void)out_size; (void)ws_size;
    const float* x     = (const float*)d_in[0];
    const float* hxs   = (const float*)d_in[1];
    const float* masks = (const float*)d_in[2];
    const float* Wih   = (const float*)d_in[3];
    const float* Whh   = (const float*)d_in[4];
    const float* bih   = (const float*)d_in[5];
    const float* bhh   = (const float*)d_in[6];
    float* out = (float*)d_out;

    char* ws = (char*)d_ws;
    size_t o = 0;
    unsigned short* xb   = (unsigned short*)(ws + o); o += (size_t)T_STEPS * N_BATCH * IN_DIM * 2;
    unsigned short* wihb = (unsigned short*)(ws + o); o += (size_t)OC_DIM * IN_DIM * 2;
    unsigned short* whhb = (unsigned short*)(ws + o); o += (size_t)OC_DIM * H_DIM * 2;
    float* gi            = (float*)(ws + o);          o += (size_t)T_STEPS * N_BATCH * OC_DIM * 4;
    float* hmf0          = (float*)(ws + o);          o += (size_t)N_BATCH * H_DIM * 4;
    float* hmf1          = (float*)(ws + o);          o += (size_t)N_BATCH * H_DIM * 4;
    unsigned short* hmb0 = (unsigned short*)(ws + o); o += (size_t)N_BATCH * H_DIM * 2;
    unsigned short* hmb1 = (unsigned short*)(ws + o);

    {
        int n = T_STEPS * N_BATCH * IN_DIM;           // 16,777,216
        cvt_bf16_kernel<<<n / 256, 256, 0, stream>>>(x, xb, n);
        n = OC_DIM * IN_DIM;                          // 3,145,728
        cvt_bf16_kernel<<<n / 256, 256, 0, stream>>>(Wih, wihb, n);
        n = OC_DIM * H_DIM;
        cvt_bf16_kernel<<<n / 256, 256, 0, stream>>>(Whh, whhb, n);
    }

    // GI for all time steps: 128 rowGroups * 48 colQuads = 6144 blocks
    gi_gemm_kernel<<<6144, 256, 0, stream>>>(xb, wihb, bih, gi);

    // hm0 = hxs * masks[0]
    init_hm_kernel<<<(N_BATCH * H_DIM) / 256, 256, 0, stream>>>(hxs, masks, hmf0, hmb0);

    // sequential recurrence: 2 rowGroups * 64 colStrips = 128 blocks per step
    float* hlast = out + (size_t)T_STEPS * N_BATCH * H_DIM;
    for (int t = 0; t < T_STEPS; ++t) {
        const float* hmf_in          = (t & 1) ? hmf1 : hmf0;
        const unsigned short* hmb_in = (t & 1) ? hmb1 : hmb0;
        float* hmf_po                = (t & 1) ? hmf0 : hmf1;
        unsigned short* hmb_po       = (t & 1) ? hmb0 : hmb1;
        gru_step_kernel<<<128, 256, 0, stream>>>(
            gi + (size_t)t * N_BATCH * OC_DIM,
            hmb_in, hmf_in, whhb, bhh,
            masks + (size_t)(t + 1) * N_BATCH,
            out + (size_t)t * N_BATCH * H_DIM,
            hlast, hmf_po, hmb_po,
            (t == T_STEPS - 1) ? 1 : 0);
    }
}
